// GATFeatureOptimizer_81896436400231
// MI455X (gfx1250) — compile-verified, matrix operand order checked by reading, not run on hardware
//
#include <hip/hip_runtime.h>
#include <hip/hip_bf16.h>

typedef __attribute__((ext_vector_type(16))) _Float16 v16h;
typedef __attribute__((ext_vector_type(8)))  float    v8f;

#define LEAKY 0.2f
#define LOG2E 1.4426950408889634f

constexpr int B_ROWS  = 8192;
constexpr int N_NODES = 2048;
constexpr int DFEAT   = 128;
constexpr int KC      = 32;             // nodes consumed per WMMA K-step
constexpr int NSTEPS  = N_NODES / KC;   // 64
constexpr int NT      = DFEAT / 16;     // 8 column tiles of 16 features
constexpr int CHUNK_BYTES = NT * KC * 16 * 2;  // 8192 B per K-chunk

// Async copy: global -> LDS, 16 bytes per lane, tracked by ASYNCcnt.
__device__ __forceinline__ void async_b128(unsigned lds_off, const void* gaddr) {
  asm volatile("global_load_async_to_lds_b128 %0, %1, off"
               :: "v"(lds_off), "v"(gaddr) : "memory");
}
__device__ __forceinline__ void wait_async0() {
  asm volatile("s_wait_asynccnt 0x0" ::: "memory");
}

// ---------------------------------------------------------------------------
// Kernel 1: per-row dot products  p[b] = P[b,:]·a_p ,  q[n] = N[n,:]·a_n
// ---------------------------------------------------------------------------
__global__ void pq_kernel(const float* __restrict__ pat,
                          const float* __restrict__ dis,
                          const float* __restrict__ attn,
                          float* __restrict__ pv, float* __restrict__ qv) {
  int wave = (blockIdx.x * blockDim.x + threadIdx.x) >> 5;
  int lane = threadIdx.x & 31;
  const float* src; const float* av; float* dst;
  if (wave < B_ROWS) {
    src = pat + (size_t)wave * DFEAT; av = attn;          dst = pv + wave;
  } else {
    int r = wave - B_ROWS;
    src = dis + (size_t)r * DFEAT;    av = attn + DFEAT;  dst = qv + r;
  }
  float4 x = *(const float4*)(src + lane * 4);
  float4 a = *(const float4*)(av  + lane * 4);
  float d = x.x*a.x + x.y*a.y + x.z*a.z + x.w*a.w;
  #pragma unroll
  for (int off = 16; off; off >>= 1) d += __shfl_xor(d, off, 32);
  if (lane == 0) *dst = d;
}

// ---------------------------------------------------------------------------
// Kernel 2: qmax = max_n q[n]  (leaky_relu monotone => row max = f(p_b+qmax))
// ---------------------------------------------------------------------------
__global__ void qmax_kernel(const float* __restrict__ qv, float* __restrict__ qm) {
  __shared__ float red[256];
  float m = -3.0e38f;
  for (int i = threadIdx.x; i < N_NODES; i += 256) m = fmaxf(m, qv[i]);
  red[threadIdx.x] = m;
  __syncthreads();
  for (int s = 128; s; s >>= 1) {
    if (threadIdx.x < s) red[threadIdx.x] = fmaxf(red[threadIdx.x], red[threadIdx.x + s]);
    __syncthreads();
  }
  if (threadIdx.x == 0) *qm = red[0];
}

// ---------------------------------------------------------------------------
// Kernel 3: pre-swizzle disease_nodes (f32, N x D) into f16 WMMA B-matrix
// fragments (ISA 7.12.2 wave32 16-bit B layout: lane = column,
// K = (lane/16)*8 + (e<8 ? e : e+8)).  One contiguous v16h per lane.
// ---------------------------------------------------------------------------
__global__ void swz_kernel(const float* __restrict__ dis, _Float16* __restrict__ swz) {
  int t  = blockIdx.x * blockDim.x + threadIdx.x;   // 64 * 8 * 32 = 16384 threads
  int kc = t >> 8;
  int L  = t & 31;
  int h  = L >> 4;
  int d  = (((t >> 5) & 7) << 4) + (L & 15);
  v16h frag;
  #pragma unroll
  for (int e = 0; e < 16; ++e) {
    int kk = h * 8 + (e < 8 ? e : e + 8);
    frag[e] = (_Float16)dis[(size_t)(kc * 32 + kk) * DFEAT + d];
  }
  *(v16h*)(swz + (size_t)t * 16) = frag;
}

// ---------------------------------------------------------------------------
// Kernel 4: fused  softmax(leaky(p+q)) @ disease  + residual.
// 4 waves / block, each wave: 16 patients x 128 features (8 f32 acc tiles).
// B chunks streamed with async global->LDS copies, double-buffered in LDS.
// ---------------------------------------------------------------------------
__global__ void __launch_bounds__(128) gat_main(
    const float* __restrict__ pat, const float* __restrict__ pv,
    const float* __restrict__ qv,  const float* __restrict__ qm,
    const _Float16* __restrict__ swz, float* __restrict__ out) {

  __shared__ float q_lds[N_NODES];                             // 8 KB
  __shared__ __align__(16) _Float16 b_lds[2][NT * KC * 16];    // 2 x 8 KB

  const int tid  = threadIdx.x;
  const int lane = tid & 31;
  const int wid  = tid >> 5;         // 0..3
  const int h    = lane >> 4;        // K-half
  const int mloc = lane & 15;        // row (A/C) / col (B/C) within tile

  for (int i = tid; i < N_NODES; i += 128) q_lds[i] = qv[i];

  const int rowbase = blockIdx.x * 64 + wid * 16;
  const float pm    = pv[rowbase + mloc];
  const float qmax  = *qm;
  const float smax  = pm + qmax;
  const float mrow  = smax > 0.f ? smax : LEAKY * smax;  // row max of leaky(scores)

  v8f acc[NT];
  #pragma unroll
  for (int t = 0; t < NT; ++t) acc[t] = (v8f){0,0,0,0,0,0,0,0};
  float sw = 0.f;

  // Each thread copies 64 B of a chunk: 4 async b128 ops.
  const char* gbase = (const char*)swz;
  const unsigned lds0 = (unsigned)(size_t)&b_lds[0][0];
  const unsigned lds1 = (unsigned)(size_t)&b_lds[1][0];

  // Prologue: start chunk 0 into buffer 0.
  #pragma unroll
  for (int j = 0; j < 4; ++j)
    async_b128(lds0 + tid * 64 + j * 16, gbase + tid * 64 + j * 16);

  for (int kc = 0; kc < NSTEPS; ++kc) {
    wait_async0();        // our async writes for chunk kc have landed
    __syncthreads();      // everyone's writes landed; prev reads of next buf done

    if (kc + 1 < NSTEPS) {
      unsigned ldsn = ((kc + 1) & 1) ? lds1 : lds0;
      const char* gn = gbase + (size_t)(kc + 1) * CHUNK_BYTES;
      #pragma unroll
      for (int j = 0; j < 4; ++j)
        async_b128(ldsn + tid * 64 + j * 16, gn + tid * 64 + j * 16);
    }

    // Build A fragment: w[m][k] = exp(leaky(p_m + q_k) - mrow), m = mloc,
    // k = h*8 + (e<8 ? e : e+8) within this 32-node chunk.
    const float* qp = &q_lds[kc * KC + h * 8];
    float4 qa = *(const float4*)(qp);
    float4 qb = *(const float4*)(qp + 4);
    float4 qc = *(const float4*)(qp + 16);
    float4 qd = *(const float4*)(qp + 20);
    float qs[16] = {qa.x,qa.y,qa.z,qa.w, qb.x,qb.y,qb.z,qb.w,
                    qc.x,qc.y,qc.z,qc.w, qd.x,qd.y,qd.z,qd.w};
    v16h afrag;
    #pragma unroll
    for (int e = 0; e < 16; ++e) {
      float s  = pm + qs[e];
      float fs = s > 0.f ? s : LEAKY * s;
      // arg <= 0 always; raw v_exp_f32 (underflow->0 is fine for softmax)
      float w  = __builtin_amdgcn_exp2f((fs - mrow) * LOG2E);
      sw += w;
      afrag[e] = (_Float16)w;
    }

    const _Float16* bbuf = b_lds[kc & 1];
    #pragma unroll
    for (int t = 0; t < NT; ++t) {
      v16h bf = *(const v16h*)&bbuf[(t * KC + lane) * 16];
      acc[t] = __builtin_amdgcn_wmma_f32_16x16x32_f16(
          false, afrag, false, bf, (short)0, acc[t], false, false);
    }
  }

  // Row sums: lane L holds the partial over its K subset; halves are disjoint.
  sw += __shfl_xor(sw, 16, 32);
  float inv = 1.f / sw;                 // valid for row m = lane & 15

  // C tile layout: VGPR r, lane L -> M = r + 8*(L/16), Ncol = L%16.
  #pragma unroll
  for (int r = 0; r < 8; ++r) {
    float invr = __shfl(inv, h * 8 + r, 32);
    int m = h * 8 + r;
    size_t base = (size_t)(rowbase + m) * DFEAT;
    #pragma unroll
    for (int t = 0; t < NT; ++t) {
      int d = t * 16 + mloc;
      out[base + d] = pat[base + d] + acc[t][r] * invr;
    }
  }
}

// ---------------------------------------------------------------------------
extern "C" void kernel_launch(void* const* d_in, const int* in_sizes, int n_in,
                              void* d_out, int out_size, void* d_ws, size_t ws_size,
                              hipStream_t stream) {
  const float* pat  = (const float*)d_in[0];   // 8192 x 128
  const float* dis  = (const float*)d_in[1];   // 2048 x 128
  const float* attn = (const float*)d_in[2];   // 256
  float* out = (float*)d_out;

  // Workspace layout (needs ~565 KB):
  char* ws = (char*)d_ws;
  _Float16* swz = (_Float16*)ws;                              // 524288 B
  float*    pv  = (float*)(ws + 524288);                      //  32768 B
  float*    qv  = (float*)(ws + 524288 + 32768);              //   8192 B
  float*    qm  = (float*)(ws + 524288 + 32768 + 8192);       //      4 B

  pq_kernel  <<<(B_ROWS + N_NODES) / 4, 128, 0, stream>>>(pat, dis, attn, pv, qv);
  qmax_kernel<<<1, 256, 0, stream>>>(qv, qm);
  swz_kernel <<<64, 256, 0, stream>>>(dis, swz);
  gat_main   <<<B_ROWS / 64, 128, 0, stream>>>(pat, pv, qv, qm, swz, out);
}
